// LlamaAttention_87600152969864
// MI455X (gfx1250) — matrix operand order repaired and checked by reading, not compiled, not run on hardware
//
#include <hip/hip_runtime.h>
#include <hip/hip_bf16.h>
#include <math.h>

// ---------------- problem constants (from reference) ----------------
#define B_   2
#define S_   2048
#define H_   4096
#define NH_  32
#define NKV_ 8
#define HD_  128
#define G_   (NH_ / NKV_)   // 4 q-heads per kv-head

typedef __attribute__((ext_vector_type(16))) _Float16 v16h;
typedef __attribute__((ext_vector_type(8)))  _Float16 v8h;
typedef __attribute__((ext_vector_type(4)))  _Float16 v4h;
typedef __attribute__((ext_vector_type(8)))  float    v8f;
typedef __attribute__((ext_vector_type(4)))  float    v4f;

static __device__ __forceinline__ v16h cat8(v8h lo, v8h hi) {
  return __builtin_shufflevector(lo, hi, 0,1,2,3,4,5,6,7,8,9,10,11,12,13,14,15);
}
static __device__ __forceinline__ v4h f2h4(v4f a) {
  v4h r; r.x=(_Float16)a.x; r.y=(_Float16)a.y; r.z=(_Float16)a.z; r.w=(_Float16)a.w;
  return r;
}

// ---- CDNA5 async copy: global -> LDS, 16B per lane, tracked by ASYNCcnt ----
static __device__ __forceinline__ void async_b128(const _Float16* g, _Float16* l) {
  unsigned lofs = (unsigned)(size_t)l;   // low 32 bits of generic ptr = LDS byte addr
  asm volatile("global_load_async_to_lds_b128 %0, %1, off"
               :: "v"(lofs), "v"(g) : "memory");
}
static __device__ __forceinline__ void wait_async0() {
  asm volatile("s_wait_asynccnt 0x0" ::: "memory");
}

// =====================================================================
// Elementwise f32 -> f16
// =====================================================================
__global__ __launch_bounds__(256) void cvt_f32_f16(const float* __restrict__ in,
                                                   _Float16* __restrict__ out, long n)
{
  long i = ((long)blockIdx.x * blockDim.x + threadIdx.x) * 4;
  if (i >= n) return;
  *(v4h*)(out + i) = f2h4(*(const v4f*)(in + i));
}

// =====================================================================
// Weight transpose + convert: in f32 [K][N] -> out f16 [N][K]
// =====================================================================
__global__ __launch_bounds__(256) void transpose_cvt(const float* __restrict__ in,
                                                     _Float16* __restrict__ out,
                                                     int K, int N)
{
  __shared__ _Float16 t[32][33];
  const int n0 = blockIdx.x * 32, k0 = blockIdx.y * 32;
#pragma unroll
  for (int j = 0; j < 4; ++j) {
    int kk = threadIdx.y + j * 8;
    t[kk][threadIdx.x] = (_Float16)in[(size_t)(k0 + kk) * N + n0 + threadIdx.x];
  }
  __syncthreads();
#pragma unroll
  for (int j = 0; j < 4; ++j) {
    int nn = threadIdx.y + j * 8;
    out[(size_t)(n0 + nn) * K + k0 + threadIdx.x] = t[threadIdx.x][nn];
  }
}

// =====================================================================
// V transpose: vh f16 [b][s][kv][hd] -> vth f16 [b][kv][hd][S]
// =====================================================================
__global__ __launch_bounds__(256) void transpose_v(const _Float16* __restrict__ in,
                                                   _Float16* __restrict__ out)
{
  __shared__ _Float16 t[32][33];
  const int bk = blockIdx.z, b = bk >> 3, kv = bk & 7;
  const int s0 = blockIdx.x * 32, d0 = blockIdx.y * 32;
#pragma unroll
  for (int j = 0; j < 4; ++j) {
    int ss = threadIdx.y + j * 8;
    t[ss][threadIdx.x] = in[((size_t)(b * S_ + s0 + ss) * NKV_ + kv) * HD_ + d0 + threadIdx.x];
  }
  __syncthreads();
#pragma unroll
  for (int j = 0; j < 4; ++j) {
    int dd = threadIdx.y + j * 8;
    out[((size_t)(b * NKV_ + kv) * HD_ + d0 + dd) * S_ + s0 + threadIdx.x] = t[threadIdx.x][dd];
  }
}

// =====================================================================
// RoPE in-place on f16 [B,S,nheads,HD] (math in f32)
// =====================================================================
__global__ void rope_f16(_Float16* __restrict__ x, const int* __restrict__ pos_ids,
                         int nheads)
{
  const int half = HD_ / 2;
  int idx = blockIdx.x * blockDim.x + threadIdx.x;
  int total = B_ * S_ * nheads * half;
  if (idx >= total) return;
  int i = idx % half;
  int t = idx / half;
  int head = t % nheads; t /= nheads;
  int s = t % S_;
  int b = t / S_;
  float pos = (float)pos_ids[b * S_ + s];
  float inv_freq = __expf(-logf(10000.0f) * (float)i * (1.0f / (float)half));
  float sn, cs;
  __sincosf(pos * inv_freq, &sn, &cs);
  size_t base = ((size_t)(b * S_ + s) * nheads + head) * HD_;
  float x1 = (float)x[base + i];
  float x2 = (float)x[base + i + half];
  x[base + i]        = (_Float16)(x1 * cs - x2 * sn);
  x[base + i + half] = (_Float16)(x2 * cs + x1 * sn);
}

// =====================================================================
// Async double-buffered GEMM: C[M,N] = A[M,K] (f16) x Bt[N,K]^T (f16)
// Tile 128x128, K-step 32, 8 waves; async-to-LDS staging (ASYNCcnt),
// ping-pong LDS, 8x v_wmma_f32_16x16x32_f16 per wave per K-step.
// =====================================================================
#define TM 128
#define TN 128
#define TK 32
#define LDT 40    // LDS row stride in halves (32 + 8 pad, 16B-aligned rows)

static __device__ __forceinline__ void stage_tile(const _Float16* __restrict__ gbase,
                                                  int ldg, int k0,
                                                  _Float16* lds, int tid)
{
  // 128 rows x 32 halves; 4x16B chunks per row; 512 chunks / 256 threads
#pragma unroll
  for (int i = 0; i < 2; ++i) {
    int idx = tid + i * 256;
    int r = idx >> 2, c = idx & 3;
    async_b128(gbase + (size_t)r * ldg + k0 + c * 8, lds + r * LDT + c * 8);
  }
}

template <typename OutT>
__global__ __launch_bounds__(256) void gemm_wmma_async(
    const _Float16* __restrict__ A, const _Float16* __restrict__ Bt,
    OutT* __restrict__ C, int M, int N, int K)
{
  __shared__ __attribute__((aligned(16))) _Float16 As[2][TM * LDT];
  __shared__ __attribute__((aligned(16))) _Float16 Bs[2][TN * LDT];

  const int tid  = threadIdx.x;
  const int lane = tid & 31;
  const int wave = tid >> 5;
  const int hi   = (lane >> 4) & 1;
  const int ln   = lane & 15;

  const int rowbase = blockIdx.y * TM;
  const int colbase = blockIdx.x * TN;
  const _Float16* Ab = A  + (size_t)rowbase * K;
  const _Float16* Bb = Bt + (size_t)colbase * K;

  v8f acc[8];
#pragma unroll
  for (int i = 0; i < 8; ++i) acc[i] = v8f{};

  // prologue: fill buffer 0
  stage_tile(Ab, K, 0, As[0], tid);
  stage_tile(Bb, K, 0, Bs[0], tid);

  const int nk = K / TK;
  for (int kt = 0; kt < nk; ++kt) {
    const int p = kt & 1;
    wait_async0();        // this wave's copies into buf p complete
    __syncthreads();      // everyone's copies done; prior compute on buf 1-p done
    if (kt + 1 < nk) {    // overlap: fill buf 1-p while computing from buf p
      stage_tile(Ab, K, (kt + 1) * TK, As[1 - p], tid);
      stage_tile(Bb, K, (kt + 1) * TK, Bs[1 - p], tid);
    }

    const int m = wave * 16 + ln;
    v8h alo = *(const v8h*)(&As[p][m * LDT + hi * 8]);
    v8h ahi = *(const v8h*)(&As[p][m * LDT + 16 + hi * 8]);
    v16h af = cat8(alo, ahi);
#pragma unroll
    for (int nt = 0; nt < 8; ++nt) {
      const int n = nt * 16 + ln;
      v8h blo = *(const v8h*)(&Bs[p][n * LDT + hi * 8]);
      v8h bhi = *(const v8h*)(&Bs[p][n * LDT + 16 + hi * 8]);
      v16h bf = cat8(blo, bhi);
      acc[nt] = __builtin_amdgcn_wmma_f32_16x16x32_f16(
          false, af, false, bf, (short)0, acc[nt], false, false);
    }
  }

  const int m0 = rowbase + wave * 16;
#pragma unroll
  for (int nt = 0; nt < 8; ++nt) {
#pragma unroll
    for (int r = 0; r < 8; ++r) {
      C[(size_t)(m0 + r + hi * 8) * N + colbase + nt * 16 + ln] = (OutT)acc[nt][r];
    }
  }
}

// =====================================================================
// Flash attention (causal, GQA): Q/K/V already f16 (V transposed [d][s]).
// Q fragments straight from global (contiguous b128 runs); K/V tiles
// async-staged into double-buffered LDS; P transposed via per-wave LDS.
// =====================================================================
#define KT  32
#define LDK 136   // Ks row stride (halves): 128 + 8 pad
#define LDV 40    // Vs row stride (halves): 32 + 8 pad
#define LDP 40    // per-wave P scratch row stride

__global__ __launch_bounds__(256) void flash_attn_wmma(
    const _Float16* __restrict__ Qh, const _Float16* __restrict__ Kh,
    const _Float16* __restrict__ Vt, _Float16* __restrict__ O)
{
  __shared__ __attribute__((aligned(16))) _Float16 Ks[2][KT * LDK];   // [t][hd]
  __shared__ __attribute__((aligned(16))) _Float16 Vs[2][HD_ * LDV];  // [d][t]
  __shared__ __attribute__((aligned(16))) _Float16 Ps[8][16 * LDP];   // per-wave

  const int tid  = threadIdx.x;
  const int lane = tid & 31;
  const int wave = tid >> 5;
  const int hi   = (lane >> 4) & 1;
  const int ln   = lane & 15;

  const int nqt  = S_ / 128;
  int bid = blockIdx.x;
  const int qt   = bid % nqt; bid /= nqt;
  const int head = bid % NH_;
  const int b    = bid / NH_;
  const int kvh  = head / G_;
  const int qblk = qt * 128;
  const int qrow0 = qblk + wave * 16;

  // ---- Q fragments directly from global f16 (two b128 runs per slice) ----
  const _Float16* qg = Qh + ((size_t)(b * S_ + qrow0 + ln) * NH_ + head) * HD_;
  v16h qf[4];
#pragma unroll
  for (int sl = 0; sl < 4; ++sl) {
    v8h lo = *(const v8h*)(qg + sl * 32 + hi * 8);
    v8h hh = *(const v8h*)(qg + sl * 32 + 16 + hi * 8);
    qf[sl] = cat8(lo, hh);
  }

  v8f oacc[8];
#pragma unroll
  for (int i = 0; i < 8; ++i) oacc[i] = v8f{};
  float rowm[8], rowl[8];
#pragma unroll
  for (int r = 0; r < 8; ++r) { rowm[r] = -INFINITY; rowl[r] = 0.0f; }

  const float scale = 0.08838834764831845f;   // 1/sqrt(128)
  const int ntile = (qblk + 128) / KT;
  const _Float16* kbase = Kh + ((size_t)(b * S_) * NKV_ + kvh) * HD_;
  const _Float16* vbase = Vt + (size_t)(b * NKV_ + kvh) * HD_ * S_;
  _Float16* Wp = &Ps[wave][0];

  // prologue: stage t-tile 0 into buffer 0
  {
#pragma unroll
    for (int i = 0; i < 2; ++i) {
      int idx = tid + i * 256;
      int kr = idx >> 4, kc = idx & 15;
      async_b128(kbase + (size_t)kr * NKV_ * HD_ + kc * 8, &Ks[0][kr * LDK + kc * 8]);
      int vr = idx >> 2, vc = idx & 3;
      async_b128(vbase + (size_t)vr * S_ + vc * 8, &Vs[0][vr * LDV + vc * 8]);
    }
  }

  for (int tb = 0; tb < ntile; ++tb) {
    const int tbase = tb * KT;
    const int p = tb & 1;
    wait_async0();
    __syncthreads();
    if (tb + 1 < ntile) {
      const int tnext = tbase + KT;
#pragma unroll
      for (int i = 0; i < 2; ++i) {
        int idx = tid + i * 256;
        int kr = idx >> 4, kc = idx & 15;
        async_b128(kbase + (size_t)(tnext + kr) * NKV_ * HD_ + kc * 8,
                   &Ks[1 - p][kr * LDK + kc * 8]);
        int vr = idx >> 2, vc = idx & 3;
        async_b128(vbase + (size_t)vr * S_ + tnext + vc * 8,
                   &Vs[1 - p][vr * LDV + vc * 8]);
      }
    }

    if (tbase <= qrow0 + 15) {   // wave has unmasked columns in this tile
      // ---- scores: two 16-col subtiles, HD reduced in 4 WMMA slices ----
      v8f s[2];
#pragma unroll
      for (int tt = 0; tt < 2; ++tt) {
        s[tt] = v8f{};
        const int tl = tt * 16 + ln;
#pragma unroll
        for (int sl = 0; sl < 4; ++sl) {
          v8h lo = *(const v8h*)(&Ks[p][tl * LDK + sl * 32 + hi * 8]);
          v8h hh = *(const v8h*)(&Ks[p][tl * LDK + sl * 32 + 16 + hi * 8]);
          v16h kf = cat8(lo, hh);
          s[tt] = __builtin_amdgcn_wmma_f32_16x16x32_f16(
              false, qf[sl], false, kf, (short)0, s[tt], false, false);
        }
      }

      // ---- scale + causal mask ----
      float pm[2][8];
#pragma unroll
      for (int tt = 0; tt < 2; ++tt) {
        const int tg = tbase + tt * 16 + ln;
#pragma unroll
        for (int r = 0; r < 8; ++r) {
          const int qg_ = qrow0 + r + hi * 8;
          float v = s[tt][r] * scale;
          pm[tt][r] = (tg <= qg_) ? v : -INFINITY;
        }
      }

      // ---- online softmax (16-lane row reductions, wave32) ----
#pragma unroll
      for (int r = 0; r < 8; ++r) {
        float mx = fmaxf(pm[0][r], pm[1][r]);
        mx = fmaxf(mx, __shfl_xor(mx, 1));
        mx = fmaxf(mx, __shfl_xor(mx, 2));
        mx = fmaxf(mx, __shfl_xor(mx, 4));
        mx = fmaxf(mx, __shfl_xor(mx, 8));
        float mnew  = fmaxf(rowm[r], mx);
        float alpha = __expf(rowm[r] - mnew);
        rowm[r] = mnew;
        rowl[r] *= alpha;
#pragma unroll
        for (int dn = 0; dn < 8; ++dn) oacc[dn][r] *= alpha;
        float rs = 0.0f;
#pragma unroll
        for (int tt = 0; tt < 2; ++tt) {
          float pv = __expf(pm[tt][r] - mnew);
          pm[tt][r] = pv;
          rs += pv;
        }
        rs += __shfl_xor(rs, 1);
        rs += __shfl_xor(rs, 2);
        rs += __shfl_xor(rs, 4);
        rs += __shfl_xor(rs, 8);
        rowl[r] += rs;
      }

      // ---- transpose P into A-fragment layout via per-wave LDS ----
#pragma unroll
      for (int tt = 0; tt < 2; ++tt) {
#pragma unroll
        for (int r = 0; r < 8; ++r) {
          Wp[(r + hi * 8) * LDP + tt * 16 + ln] = (_Float16)pm[tt][r];
        }
      }
      v8h plo = *(const v8h*)(&Wp[ln * LDP + hi * 8]);
      v8h phi = *(const v8h*)(&Wp[ln * LDP + 16 + hi * 8]);
      v16h pf = cat8(plo, phi);

      // ---- PV: 8 WMMA across HD ----
#pragma unroll
      for (int dn = 0; dn < 8; ++dn) {
        const int d = dn * 16 + ln;
        v8h vlo = *(const v8h*)(&Vs[p][d * LDV + hi * 8]);
        v8h vhi = *(const v8h*)(&Vs[p][d * LDV + 16 + hi * 8]);
        v16h vf = cat8(vlo, vhi);
        oacc[dn] = __builtin_amdgcn_wmma_f32_16x16x32_f16(
            false, pf, false, vf, (short)0, oacc[dn], false, false);
      }
    }
  }

  // ---- normalize and store O (f16): [b, s, head, d] ----
#pragma unroll
  for (int r = 0; r < 8; ++r) {
    float inv = 1.0f / rowl[r];
    const int qg_ = qrow0 + r + hi * 8;
    _Float16* og = O + ((size_t)(b * S_ + qg_) * NH_ + head) * HD_;
#pragma unroll
    for (int dn = 0; dn < 8; ++dn) og[dn * 16 + ln] = (_Float16)(oacc[dn][r] * inv);
  }
}

// =====================================================================
// Host-side launch
// =====================================================================
extern "C" void kernel_launch(void* const* d_in, const int* in_sizes, int n_in,
                              void* d_out, int out_size, void* d_ws, size_t ws_size,
                              hipStream_t stream) {
  (void)in_sizes; (void)n_in; (void)out_size; (void)ws_size;

  const float* hidden = (const float*)d_in[0];
  const float* wq     = (const float*)d_in[1];
  const float* wk     = (const float*)d_in[2];
  const float* wv     = (const float*)d_in[3];
  const float* wo     = (const float*)d_in[4];
  const int*   pos    = (const int*)d_in[5];
  float* out = (float*)d_out;

  char* ws = (char*)d_ws;                                   // 152 MB total
  _Float16* ah  = (_Float16*)(ws);                          // 32 MB: hidden f16
  _Float16* wt  = (_Float16*)(ws + ((size_t)32  << 20));    // 32 MB: transposed weight (reused)
  _Float16* qh  = (_Float16*)(ws + ((size_t)64  << 20));    // 32 MB
  _Float16* kh  = (_Float16*)(ws + ((size_t)96  << 20));    //  8 MB
  _Float16* vh  = (_Float16*)(ws + ((size_t)104 << 20));    //  8 MB
  _Float16* vth = (_Float16*)(ws + ((size_t)112 << 20));    //  8 MB
  _Float16* oh  = (_Float16*)(ws + ((size_t)120 << 20));    // 32 MB

  const int M = B_ * S_;            // 4096
  dim3 blk(256), tb(32, 8);
  long nh_elems = (long)M * H_;

  // hidden -> f16
  cvt_f32_f16<<<(unsigned)(nh_elems / 4 / 256), blk, 0, stream>>>(hidden, ah, nh_elems);

  // Q projection
  transpose_cvt<<<dim3((NH_ * HD_) / 32, H_ / 32), tb, 0, stream>>>(wq, wt, H_, NH_ * HD_);
  gemm_wmma_async<_Float16><<<dim3((NH_ * HD_) / TN, M / TM), blk, 0, stream>>>(
      ah, wt, qh, M, NH_ * HD_, H_);
  // K projection
  transpose_cvt<<<dim3((NKV_ * HD_) / 32, H_ / 32), tb, 0, stream>>>(wk, wt, H_, NKV_ * HD_);
  gemm_wmma_async<_Float16><<<dim3((NKV_ * HD_) / TN, M / TM), blk, 0, stream>>>(
      ah, wt, kh, M, NKV_ * HD_, H_);
  // V projection
  transpose_cvt<<<dim3((NKV_ * HD_) / 32, H_ / 32), tb, 0, stream>>>(wv, wt, H_, NKV_ * HD_);
  gemm_wmma_async<_Float16><<<dim3((NKV_ * HD_) / TN, M / TM), blk, 0, stream>>>(
      ah, wt, vh, M, NKV_ * HD_, H_);

  // RoPE on q,k (in-place f16)
  int nq = B_ * S_ * NH_ * (HD_ / 2);
  rope_f16<<<(nq + 255) / 256, 256, 0, stream>>>(qh, pos, NH_);
  int nk = B_ * S_ * NKV_ * (HD_ / 2);
  rope_f16<<<(nk + 255) / 256, 256, 0, stream>>>(kh, pos, NKV_);

  // V -> [b][kv][hd][s] for PV fragments
  transpose_v<<<dim3(S_ / 32, HD_ / 32, B_ * NKV_), tb, 0, stream>>>(vh, vth);

  // causal GQA flash attention -> oh (f16)
  flash_attn_wmma<<<B_ * NH_ * (S_ / 128), blk, 0, stream>>>(qh, kh, vth, oh);

  // output projection -> d_out (f32)
  transpose_cvt<<<dim3(H_ / 32, H_ / 32), tb, 0, stream>>>(wo, wt, H_, H_);
  gemm_wmma_async<float><<<dim3(H_ / TN, M / TM), blk, 0, stream>>>(
      oh, wt, out, M, H_, H_);
}
